// PatchDiscriminator_76785425318140
// MI455X (gfx1250) — matrix operand order repaired and checked by reading, not compiled, not run on hardware
//
#include <hip/hip_runtime.h>
#include <hip/hip_bf16.h>
#include <hip/hip_fp16.h>

typedef __attribute__((ext_vector_type(16))) _Float16 v16h;
typedef __attribute__((ext_vector_type(8)))  _Float16 v8h;
typedef __attribute__((ext_vector_type(8)))  float    v8f;
typedef __attribute__((ext_vector_type(4)))  float    v4f;
typedef __attribute__((ext_vector_type(4)))  unsigned int u32x4;
typedef __attribute__((ext_vector_type(8)))  int      i32x8;
typedef __attribute__((ext_vector_type(4)))  int      i32x4;

#define BATCH   16
#define SEQL    8192
#define VOCAB   512
#define DIMD    512
#define NLAYERS 8
#define NHEADS  8
#define HD      64
#define FFD     2048
#define PATCH   16
#define SS      513            // 512 patches + cls
#define ROWS    (BATCH * SS)   // 8208

#if __has_builtin(__builtin_amdgcn_tensor_load_to_lds) && __has_builtin(__builtin_amdgcn_s_wait_tensorcnt)
#define USE_TDM 1
#else
#define USE_TDM 0
#endif

// ---------------------------------------------------------------------------
// WMMA GEMM: C[M,N] = act(A[M,K](f32,lda) @ W(f16, pre-transposed [N][K]) + bias)
// Block tile 64(M) x 128(N) x 64(K), 8 waves, double-buffered LDS.
// B tiles fetched by the Tensor Data Mover (async, LDS-padded rows); A tiles
// converted f32->f16 through registers. One barrier per 8 WMMAs.
// ---------------------------------------------------------------------------
#define BM 64
#define BN 128
#define BK 64
#define LDP (BK + 8)           // 72 halfs = 144B row stride (conflict-free)

__device__ __forceinline__ float gelu_tanh(float x) {
    float x3 = x * x * x;
    return 0.5f * x * (1.0f + tanhf(0.7978845608028654f * (x + 0.044715f * x3)));
}

#if USE_TDM
// Issue one TDM load of a [BN rows x BK halfs] tile of WT (row stride K halfs)
// into LDS at ldsOff, padding +16B per 128B row => LDS row stride 144B.
__device__ __forceinline__ void tdm_load_b_tile(const _Float16* gsrc, unsigned ldsOff, int K) {
    unsigned long long ga = (unsigned long long)(size_t)gsrc;
    u32x4 g0;
    g0[0] = 1u;                                   // count=1 (valid), user mode
    g0[1] = ldsOff;                               // LDS byte address
    g0[2] = (unsigned)ga;                         // global addr [31:0]
    g0[3] = ((unsigned)(ga >> 32) & 0x01FFFFFFu)  // global addr [56:32]
          | (2u << 30);                           // type = 2 ("image")
    unsigned long long q0 = (1ull << 16)          // data_size = 2 bytes
                          | (1ull << 20)          // pad_enable
                          | (4ull << 22)          // pad_interval = 32 DW (128B)
                          | (3ull << 25)          // pad_amount   = 4 DW (16B)
                          | (64ull << 48);        // tensor_dim0 = 64
    unsigned long long q1 = (128ull << 16)        // tensor_dim1 = 128
                          | (64ull << 48);        // tile_dim0 = 64
    unsigned long long q2 = 128ull                // tile_dim1 = 128 (tile_dim2=0)
                          | ((unsigned long long)(unsigned)K << 32); // dim0 stride
    unsigned long long q3 = 0ull;
    i32x8 g1;
    g1[0] = (int)(unsigned)q0; g1[1] = (int)(q0 >> 32);
    g1[2] = (int)(unsigned)q1; g1[3] = (int)(q1 >> 32);
    g1[4] = (int)(unsigned)q2; g1[5] = (int)(q2 >> 32);
    g1[6] = (int)(unsigned)q3; g1[7] = (int)(q3 >> 32);
    i32x4 z4 = {0, 0, 0, 0};
    i32x8 z8 = {0, 0, 0, 0, 0, 0, 0, 0};
    __builtin_amdgcn_tensor_load_to_lds(g0, g1, z4, z4, z8, 0);
}
#endif

__global__ __launch_bounds__(256) void gemm_f16w(
    const float* __restrict__ A, int lda, long aBatch,
    const _Float16* __restrict__ WT, long wBatch,    // [N][K] (transposed)
    const float* __restrict__ bias,
    float* __restrict__ C, int ldc, long cBatch,
    int M, int N, int K, int fuse /*0 none, 1 gelu*/)
{
    __shared__ __align__(16) _Float16 sA[2][BM][LDP];
    __shared__ __align__(16) _Float16 sB[2][BN][LDP];

    A  += (long)blockIdx.z * aBatch;
    WT += (long)blockIdx.z * wBatch;
    C  += (long)blockIdx.z * cBatch;

    const int tid  = threadIdx.x;
    const int lane = tid & 31;
    const int wave = tid >> 5;
    const int wm   = wave & 3;
    const int wn   = wave >> 2;
    const int bm   = blockIdx.y * BM;
    const int bn   = blockIdx.x * BN;

    // A staging: row = tid>>2, k-chunk (tid&3)*16 (16 floats -> 16 halfs)
    const int ar = tid >> 2;
    const int ak = (tid & 3) * 16;
    long arow = (long)bm + ar;
    if (arow > (long)M - 1) arow = (long)M - 1;
    const float* aptr = A + arow * (long)lda + ak;

    const _Float16* wtile = WT + (long)bn * K;      // B tile origin (row 0)
#if !USE_TDM
    // manual B staging: row = tid>>1, 32-half chunk (tid&1)*32
    const _Float16* bptr = wtile + (long)(tid >> 1) * K + (tid & 1) * 32;
#endif

    v8f acc[4] = {v8f{}, v8f{}, v8f{}, v8f{}};

    // WMMA fragment addressing (ISA 16-bit A 16x32 / B 32x16 layouts)
    const int frow = wm * 16 + (lane & 15);
    const int kbA  = (lane < 16) ? 0 : 8;
    const int fcol = wn * 64 + (lane & 15);
    const int kbB  = (lane < 16) ? 0 : 8;

    const unsigned ldsB0 = (unsigned)(size_t)&sB[0][0][0];
    const unsigned ldsB1 = (unsigned)(size_t)&sB[1][0][0];

    // ---- prologue: stage 0 ----
    {
        v4f a0 = *(const v4f*)(aptr);
        v4f a1 = *(const v4f*)(aptr + 4);
        v4f a2 = *(const v4f*)(aptr + 8);
        v4f a3 = *(const v4f*)(aptr + 12);
#if USE_TDM
        if (tid < 32) tdm_load_b_tile(wtile, ldsB0, K);
#else
        v8h b0 = *(const v8h*)(bptr);
        v8h b1 = *(const v8h*)(bptr + 8);
        v8h b2 = *(const v8h*)(bptr + 16);
        v8h b3 = *(const v8h*)(bptr + 24);
#endif
        v8h h0, h1;
#pragma unroll
        for (int i = 0; i < 4; ++i) {
            h0[i] = (_Float16)a0[i]; h0[4 + i] = (_Float16)a1[i];
            h1[i] = (_Float16)a2[i]; h1[4 + i] = (_Float16)a3[i];
        }
        *(v8h*)&sA[0][ar][ak]     = h0;
        *(v8h*)&sA[0][ar][ak + 8] = h1;
#if USE_TDM
        if (tid < 32) __builtin_amdgcn_s_wait_tensorcnt((short)0);
#else
        {
            const int brow = tid >> 1, bkc = (tid & 1) * 32;
            *(v8h*)&sB[0][brow][bkc]      = b0;
            *(v8h*)&sB[0][brow][bkc + 8]  = b1;
            *(v8h*)&sB[0][brow][bkc + 16] = b2;
            *(v8h*)&sB[0][brow][bkc + 24] = b3;
        }
#endif
        __syncthreads();
    }

    int cur = 0;
    for (int k0 = 0; k0 < K; k0 += BK, cur ^= 1) {
        const bool hasNext = (k0 + BK) < K;
        v4f a0, a1, a2, a3;
#if !USE_TDM
        v8h b0, b1, b2, b3;
#endif
        if (hasNext) {
            a0 = *(const v4f*)(aptr + BK);
            a1 = *(const v4f*)(aptr + BK + 4);
            a2 = *(const v4f*)(aptr + BK + 8);
            a3 = *(const v4f*)(aptr + BK + 12);
#if USE_TDM
            if (tid < 32) tdm_load_b_tile(wtile + k0 + BK, (cur ? ldsB0 : ldsB1), K);
#else
            b0 = *(const v8h*)(bptr + k0 + BK);
            b1 = *(const v8h*)(bptr + k0 + BK + 8);
            b2 = *(const v8h*)(bptr + k0 + BK + 16);
            b3 = *(const v8h*)(bptr + k0 + BK + 24);
#endif
            if (k0 + 2 * BK < K) __builtin_prefetch(aptr + 2 * BK, 0, 1);
        }

        // ---- compute current stage: 2 k-substeps x 4 N-tiles = 8 WMMAs ----
#pragma unroll
        for (int kk = 0; kk < BK; kk += 32) {
            v16h af;
            ((v8h*)&af)[0] = *(const v8h*)&sA[cur][frow][kk + kbA];
            ((v8h*)&af)[1] = *(const v8h*)&sA[cur][frow][kk + kbA + 16];
#pragma unroll
            for (int j = 0; j < 4; ++j) {
                v16h bf;
                ((v8h*)&bf)[0] = *(const v8h*)&sB[cur][fcol + j * 16][kk + kbB];
                ((v8h*)&bf)[1] = *(const v8h*)&sB[cur][fcol + j * 16][kk + kbB + 16];
                acc[j] = __builtin_amdgcn_wmma_f32_16x16x32_f16(
                    false, af, false, bf, (short)0, acc[j], false, false);
            }
        }

        if (hasNext) {
            const int nxt = cur ^ 1;
            v8h h0, h1;
#pragma unroll
            for (int i = 0; i < 4; ++i) {
                h0[i] = (_Float16)a0[i]; h0[4 + i] = (_Float16)a1[i];
                h1[i] = (_Float16)a2[i]; h1[4 + i] = (_Float16)a3[i];
            }
            *(v8h*)&sA[nxt][ar][ak]     = h0;
            *(v8h*)&sA[nxt][ar][ak + 8] = h1;
#if USE_TDM
            if (tid < 32) __builtin_amdgcn_s_wait_tensorcnt((short)0);
#else
            {
                const int brow = tid >> 1, bkc = (tid & 1) * 32;
                *(v8h*)&sB[nxt][brow][bkc]      = b0;
                *(v8h*)&sB[nxt][brow][bkc + 8]  = b1;
                *(v8h*)&sB[nxt][brow][bkc + 16] = b2;
                *(v8h*)&sB[nxt][brow][bkc + 24] = b3;
            }
#endif
            __syncthreads();
            aptr += BK;
        }
    }

    // ---- epilogue: C/D layout = VGPR r: lanes 0-15 M=r, lanes 16-31 M=r+8 ----
    const int mbase = bm + wm * 16 + ((lane >> 4) * 8);
    const int nbase = bn + wn * 64 + (lane & 15);
#pragma unroll
    for (int j = 0; j < 4; ++j) {
        const int gn = nbase + j * 16;
        const float bv = bias ? bias[gn] : 0.0f;
#pragma unroll
        for (int r = 0; r < 8; ++r) {
            const int gm = mbase + r;
            if (gm < M) {
                float v = acc[j][r] + bv;
                if (fuse == 1) v = gelu_tanh(v);
                C[(long)gm * ldc + gn] = v;
            }
        }
    }
}

// ---------------------------------------------------------------------------
// Prep kernels
// ---------------------------------------------------------------------------
// dst[N][K] (f16) = transpose(src[K][N] (f32)); batched over blockIdx.z
__global__ __launch_bounds__(256) void transpose_f16t_k(
    const float* __restrict__ src, long sBatch,
    _Float16* __restrict__ dst, long dBatch, int K, int N)
{
    __shared__ _Float16 tile[32][33];
    src += (long)blockIdx.z * sBatch;
    dst += (long)blockIdx.z * dBatch;
    const int n0 = blockIdx.x * 32, k0 = blockIdx.y * 32;
    const int tx = threadIdx.x & 31, ty = threadIdx.x >> 5;
#pragma unroll
    for (int r = 0; r < 4; ++r) {
        const int k = ty + r * 8;
        tile[k][tx] = (_Float16)src[(long)(k0 + k) * N + n0 + tx];
    }
    __syncthreads();
#pragma unroll
    for (int r = 0; r < 4; ++r) {
        const int n = ty + r * 8;
        dst[(long)(n0 + n) * K + k0 + tx] = tile[tx][n];
    }
}

// ctT[(k*512 + o)*512 + i] = conv_w[(o*512 + i)*16 + k]   (f16, "WT" per k)
__global__ void convw_to_f16t_k(const float* __restrict__ cw, _Float16* __restrict__ ct) {
    long idx = (long)blockIdx.x * blockDim.x + threadIdx.x;
    if (idx >= (long)PATCH * DIMD * DIMD) return;
    const int i = (int)(idx & 511);
    const int o = (int)((idx >> 9) & 511);
    const int k = (int)(idx >> 18);
    ct[idx] = (_Float16)cw[((long)o * DIMD + i) * PATCH + k];
}

// ebias[o] = conv_b[o] + sum_i embed_b[i] * sum_k conv_w[o,i,k]
__global__ __launch_bounds__(256) void ebias_k(const float* __restrict__ cw,
                                               const float* __restrict__ cb,
                                               const float* __restrict__ eb,
                                               float* __restrict__ ebias) {
    const int o = blockIdx.x, t = threadIdx.x;
    __shared__ float red[256];
    float s = 0.0f;
    for (int i = t; i < DIMD; i += 256) {
        float cs = 0.0f;
#pragma unroll
        for (int k = 0; k < PATCH; ++k) cs += cw[((long)o * DIMD + i) * PATCH + k];
        s += eb[i] * cs;
    }
    red[t] = s; __syncthreads();
    for (int off = 128; off > 0; off >>= 1) { if (t < off) red[t] += red[t + off]; __syncthreads(); }
    if (t == 0) ebias[o] = cb[o] + red[0];
}

// ---------------------------------------------------------------------------
// Elementwise kernels
// ---------------------------------------------------------------------------
__global__ void assemble_x_k(const float* __restrict__ Y, const float* __restrict__ ebias,
                             const float* __restrict__ cls, const float* __restrict__ pos,
                             float* __restrict__ x) {
    const int s = blockIdx.x, b = blockIdx.y, t = threadIdx.x;
#pragma unroll
    for (int r = 0; r < 2; ++r) {
        const int c = t + r * 256;
        float v;
        if (s == 0) v = cls[c] + pos[c];
        else        v = Y[((long)b * 512 + (s - 1)) * DIMD + c] + ebias[c] + pos[(long)s * DIMD + c];
        x[((long)b * SS + s) * DIMD + c] = v;
    }
}

__device__ __forceinline__ float elu1(float x) { return x > 0.0f ? x + 1.0f : __expf(x); }

__global__ __launch_bounds__(256) void rotary_elu_k(float* __restrict__ q, float* __restrict__ k) {
    const int row = blockIdx.x;
    const int s   = row % SS;
    const int t   = threadIdx.x;
    const int h = t >> 5, j = t & 31;
    const float inv = __powf(10000.0f, -(float)j / 32.0f);
    float sn, cs;
    __sincosf((float)s * inv, &sn, &cs);
    const long base = (long)row * DIMD + h * HD;

    float x1 = q[base + j], x2 = q[base + j + 32];
    q[base + j]      = elu1(x1 * cs - x2 * sn);
    q[base + j + 32] = elu1(x2 * cs + x1 * sn);

    x1 = k[base + j]; x2 = k[base + j + 32];
    k[base + j]      = elu1(x1 * cs - x2 * sn);
    k[base + j + 32] = elu1(x2 * cs + x1 * sn);
}

__global__ __launch_bounds__(256) void kv_reduce_k(const float* __restrict__ k,
                                                   const float* __restrict__ v,
                                                   float* __restrict__ kv,
                                                   float* __restrict__ ksum) {
    const int h = blockIdx.x, b = blockIdx.y, t = threadIdx.x;
    __shared__ float sk[4][64], sv[4][64];
    const int d  = t >> 2;
    const int mb = (t & 3) * 16;
    float acc[16];
#pragma unroll
    for (int m = 0; m < 16; ++m) acc[m] = 0.0f;
    float ks = 0.0f;

    for (int s0 = 0; s0 < 516; s0 += 4) {
        int li = t;
#pragma unroll
        for (int rep = 0; rep < 2; ++rep, li += 256) {
            const int sr = li >> 7;
            const int cc = li & 127;
            const int s  = s0 + sr;
            float val = 0.0f;
            if (s < SS) {
                const long base = ((long)b * SS + s) * DIMD + h * HD;
                val = (cc < 64) ? k[base + cc] : v[base + (cc - 64)];
            }
            if (cc < 64) sk[sr][cc] = val; else sv[sr][cc - 64] = val;
        }
        __syncthreads();
#pragma unroll
        for (int sr = 0; sr < 4; ++sr) {
            const float kd = sk[sr][d];
            if ((t & 3) == 0) ks += kd;
#pragma unroll
            for (int m = 0; m < 16; ++m) acc[m] += kd * sv[sr][mb + m];
        }
        __syncthreads();
    }
    const long o = (((long)b * NHEADS + h) * HD + d) * HD + mb;
#pragma unroll
    for (int m = 0; m < 16; ++m) kv[o + m] = acc[m];
    if ((t & 3) == 0) ksum[((long)b * NHEADS + h) * HD + d] = ks;
}

__global__ __launch_bounds__(256) void attn_apply_k(const float* __restrict__ q,
                                                    const float* __restrict__ kv,
                                                    const float* __restrict__ ksum,
                                                    float* __restrict__ out) {
    const int row = blockIdx.x;
    const int b   = row / SS;
    const int t   = threadIdx.x;
    __shared__ float sq[512];
    sq[t]       = q[(long)row * DIMD + t];
    sq[t + 256] = q[(long)row * DIMD + t + 256];
    __syncthreads();
#pragma unroll
    for (int rep = 0; rep < 2; ++rep) {
        const int idx = t + rep * 256;
        const int h = idx >> 6, m = idx & 63;
        const float* kvh = kv + (((long)b * NHEADS + h) * HD) * HD;
        const float* ksh = ksum + ((long)b * NHEADS + h) * HD;
        const float* qh  = sq + h * HD;
        float dz = 0.0f, o = 0.0f;
#pragma unroll 8
        for (int d = 0; d < HD; ++d) {
            const float qd = qh[d];
            dz += qd * ksh[d];
            o  += qd * kvh[d * HD + m];
        }
        out[(long)row * DIMD + idx] = o / (dz + 1e-6f);
    }
}

__global__ __launch_bounds__(256) void ln_rows_k(const float* __restrict__ X, long xstride,
                                                 const float* __restrict__ R,
                                                 const float* __restrict__ g,
                                                 const float* __restrict__ bb,
                                                 float* __restrict__ out, long ostride) {
    const long row = blockIdx.x;
    const int  t   = threadIdx.x;
    const float* xr = X + row * xstride;
    float v0 = xr[t]       + (R ? R[row * DIMD + t]       : 0.0f);
    float v1 = xr[t + 256] + (R ? R[row * DIMD + t + 256] : 0.0f);
    __shared__ float red[256];
    red[t] = v0 + v1; __syncthreads();
    for (int off = 128; off > 0; off >>= 1) { if (t < off) red[t] += red[t + off]; __syncthreads(); }
    const float mean = red[0] * (1.0f / DIMD);
    __syncthreads();
    const float d0 = v0 - mean, d1 = v1 - mean;
    red[t] = d0 * d0 + d1 * d1; __syncthreads();
    for (int off = 128; off > 0; off >>= 1) { if (t < off) red[t] += red[t + off]; __syncthreads(); }
    const float rstd = rsqrtf(red[0] * (1.0f / DIMD) + 1e-5f);
    out[row * ostride + t]       = d0 * rstd * g[t]       + bb[t];
    out[row * ostride + t + 256] = d1 * rstd * g[t + 256] + bb[t + 256];
}

// ---------------------------------------------------------------------------
// Host orchestration
// ---------------------------------------------------------------------------
static inline void launch_gemm(const float* A, int lda, long aB,
                               const _Float16* WT, long wB, const float* bias,
                               float* C, int ldc, long cB,
                               int M, int N, int K, int fuse, int batch, hipStream_t st) {
    dim3 grid(N / BN, (M + BM - 1) / BM, batch);
    gemm_f16w<<<grid, 256, 0, st>>>(A, lda, aB, WT, wB, bias, C, ldc, cB, M, N, K, fuse);
}

extern "C" void kernel_launch(void* const* d_in, const int* in_sizes, int n_in,
                              void* d_out, int out_size, void* d_ws, size_t ws_size,
                              hipStream_t stream) {
    (void)in_sizes; (void)n_in; (void)out_size; (void)ws_size;
    const float* inputs  = (const float*)d_in[0];
    const float* embed_w = (const float*)d_in[1];
    const float* embed_b = (const float*)d_in[2];
    const float* conv_w  = (const float*)d_in[3];
    const float* conv_b  = (const float*)d_in[4];
    const float* pos_emb = (const float*)d_in[5];
    const float* clsv    = (const float*)d_in[6];
    const float* wq = (const float*)d_in[7];   const float* bq = (const float*)d_in[8];
    const float* wk = (const float*)d_in[9];   const float* bk = (const float*)d_in[10];
    const float* wv = (const float*)d_in[11];  const float* bv = (const float*)d_in[12];
    const float* wo = (const float*)d_in[13];  const float* bo = (const float*)d_in[14];
    const float* ln1g = (const float*)d_in[15]; const float* ln1b = (const float*)d_in[16];
    const float* w1 = (const float*)d_in[17];  const float* b1 = (const float*)d_in[18];
    const float* w2 = (const float*)d_in[19];  const float* b2 = (const float*)d_in[20];
    const float* ln2g = (const float*)d_in[21]; const float* ln2b = (const float*)d_in[22];
    const float* lnfg = (const float*)d_in[23]; const float* lnfb = (const float*)d_in[24];
    const float* out_w = (const float*)d_in[25]; const float* out_b = (const float*)d_in[26];

    char* wsb = (char*)d_ws;
    size_t off = 0;
    auto carve = [&](size_t bytes) -> void* {
        void* p = wsb + off; off += (bytes + 255) & ~(size_t)255; return p;
    };
    const long WSZ  = (long)NLAYERS * DIMD * DIMD;
    const long W1SZ = (long)NLAYERS * DIMD * FFD;
    _Float16* wqT = (_Float16*)carve(WSZ * 2);       // [l][512 n][512 k]
    _Float16* wkT = (_Float16*)carve(WSZ * 2);
    _Float16* wvT = (_Float16*)carve(WSZ * 2);
    _Float16* woT = (_Float16*)carve(WSZ * 2);
    _Float16* w1T = (_Float16*)carve(W1SZ * 2);      // [l][2048 n][512 k]
    _Float16* w2T = (_Float16*)carve(W1SZ * 2);      // [l][512 n][2048 k]
    _Float16* owT = (_Float16*)carve((long)DIMD * VOCAB * 2);
    _Float16* ctT = (_Float16*)carve((long)PATCH * DIMD * DIMD * 2);  // [k][o][i]
    _Float16* wcT = (_Float16*)carve((long)DIMD * PATCH * VOCAB * 2); // [o][8192]
    float* wcf32  = (float*)carve((long)PATCH * VOCAB * DIMD * 4);    // [k][v][o]
    float* ebias  = (float*)carve(DIMD * 4);
    float* Y      = (float*)carve((long)BATCH * 512 * DIMD * 4);
    float* x      = (float*)carve((long)ROWS * DIMD * 4);
    float* qb     = (float*)carve((long)ROWS * DIMD * 4);
    float* kb     = (float*)carve((long)ROWS * DIMD * 4);
    float* vb     = (float*)carve((long)ROWS * DIMD * 4);
    float* ab     = (float*)carve((long)ROWS * DIMD * 4);
    float* ob     = (float*)carve((long)ROWS * DIMD * 4);
    float* ffb    = (float*)carve((long)ROWS * FFD * 4);
    float* kvb    = (float*)carve((long)BATCH * NHEADS * HD * HD * 4);
    float* ksb    = (float*)carve((long)BATCH * NHEADS * HD * 4);
    float* clsb   = (float*)carve((long)BATCH * DIMD * 4);

    // --- weight prep: transpose + f32->f16 ---
    transpose_f16t_k<<<dim3(DIMD/32, DIMD/32, NLAYERS), 256, 0, stream>>>(wq, (long)DIMD*DIMD, wqT, (long)DIMD*DIMD, DIMD, DIMD);
    transpose_f16t_k<<<dim3(DIMD/32, DIMD/32, NLAYERS), 256, 0, stream>>>(wk, (long)DIMD*DIMD, wkT, (long)DIMD*DIMD, DIMD, DIMD);
    transpose_f16t_k<<<dim3(DIMD/32, DIMD/32, NLAYERS), 256, 0, stream>>>(wv, (long)DIMD*DIMD, wvT, (long)DIMD*DIMD, DIMD, DIMD);
    transpose_f16t_k<<<dim3(DIMD/32, DIMD/32, NLAYERS), 256, 0, stream>>>(wo, (long)DIMD*DIMD, woT, (long)DIMD*DIMD, DIMD, DIMD);
    transpose_f16t_k<<<dim3(FFD/32,  DIMD/32, NLAYERS), 256, 0, stream>>>(w1, (long)DIMD*FFD,  w1T, (long)DIMD*FFD,  DIMD, FFD);
    transpose_f16t_k<<<dim3(DIMD/32, FFD/32,  NLAYERS), 256, 0, stream>>>(w2, (long)FFD*DIMD,  w2T, (long)FFD*DIMD,  FFD, DIMD);
    transpose_f16t_k<<<dim3(DIMD/32, DIMD/32, 1), 256, 0, stream>>>(out_w, 0, owT, 0, DIMD, VOCAB);
    convw_to_f16t_k<<<dim3((PATCH * DIMD * DIMD + 255) / 256), 256, 0, stream>>>(conv_w, ctT);
    ebias_k<<<dim3(DIMD), 256, 0, stream>>>(conv_w, conv_b, embed_b, ebias);

    // --- fused embed+conv weight: Wc[k][v][o] = embed_w @ ct[k]  (batched z=16) ---
    launch_gemm(embed_w, DIMD, 0, ctT, (long)DIMD * DIMD, nullptr,
                wcf32, DIMD, (long)DIMD * DIMD, DIMD, DIMD, DIMD, 0, PATCH, stream);
    // transpose [8192][512] -> wcT [512][8192]
    transpose_f16t_k<<<dim3(DIMD/32, (PATCH*VOCAB)/32, 1), 256, 0, stream>>>(
        wcf32, 0, wcT, 0, PATCH * VOCAB, DIMD);

    // --- patch embedding: Y[8192,512] = inputs_[8192,8192] @ Wc ---
    launch_gemm(inputs, PATCH * VOCAB, 0, wcT, 0, nullptr,
                Y, DIMD, 0, BATCH * 512, DIMD, PATCH * VOCAB, 0, 1, stream);
    assemble_x_k<<<dim3(SS, BATCH), 256, 0, stream>>>(Y, ebias, clsv, pos_emb, x);

    // --- transformer layers ---
    for (int l = 0; l < NLAYERS; ++l) {
        const _Float16* wql = wqT + (long)l * DIMD * DIMD;
        const _Float16* wkl = wkT + (long)l * DIMD * DIMD;
        const _Float16* wvl = wvT + (long)l * DIMD * DIMD;
        const _Float16* wol = woT + (long)l * DIMD * DIMD;
        const _Float16* w1l = w1T + (long)l * DIMD * FFD;
        const _Float16* w2l = w2T + (long)l * FFD * DIMD;

        launch_gemm(x, DIMD, 0, wql, 0, bq + (long)l * DIMD, qb, DIMD, 0, ROWS, DIMD, DIMD, 0, 1, stream);
        launch_gemm(x, DIMD, 0, wkl, 0, bk + (long)l * DIMD, kb, DIMD, 0, ROWS, DIMD, DIMD, 0, 1, stream);
        launch_gemm(x, DIMD, 0, wvl, 0, bv + (long)l * DIMD, vb, DIMD, 0, ROWS, DIMD, DIMD, 0, 1, stream);

        rotary_elu_k<<<dim3(ROWS), 256, 0, stream>>>(qb, kb);
        kv_reduce_k<<<dim3(NHEADS, BATCH), 256, 0, stream>>>(kb, vb, kvb, ksb);
        attn_apply_k<<<dim3(ROWS), 256, 0, stream>>>(qb, kvb, ksb, ab);

        launch_gemm(ab, DIMD, 0, wol, 0, bo + (long)l * DIMD, ob, DIMD, 0, ROWS, DIMD, DIMD, 0, 1, stream);
        ln_rows_k<<<dim3(ROWS), 256, 0, stream>>>(x, DIMD, ob, ln1g + (long)l * DIMD,
                                                  ln1b + (long)l * DIMD, x, DIMD);

        launch_gemm(x, DIMD, 0, w1l, 0, b1 + (long)l * FFD, ffb, FFD, 0, ROWS, FFD, DIMD, 1, 1, stream);
        launch_gemm(ffb, FFD, 0, w2l, 0, b2 + (long)l * DIMD, ob, DIMD, 0, ROWS, DIMD, FFD, 0, 1, stream);
        ln_rows_k<<<dim3(ROWS), 256, 0, stream>>>(x, DIMD, ob, ln2g + (long)l * DIMD,
                                                  ln2b + (long)l * DIMD, x, DIMD);
    }

    // --- final LN on cls rows + output projection ---
    ln_rows_k<<<dim3(BATCH), 256, 0, stream>>>(x, (long)SS * DIMD, nullptr, lnfg, lnfb, clsb, DIMD);
    launch_gemm(clsb, DIMD, 0, owT, 0, out_b, (float*)d_out, VOCAB, 0,
                BATCH, VOCAB, DIMD, 0, 1, stream);
}
